// Mamba_33646773796908
// MI455X (gfx1250) — compile-verified, matrix operand order checked by reading, not run on hardware
//
#include <hip/hip_runtime.h>
#include <hip/hip_bf16.h>
#include <stdint.h>

// ---------------------------------------------------------------------------
// Types for CDNA5 WMMA (wave32): bf16 A/B fragments (16 elems = 8 VGPRs),
// f32 accumulator (8 elems = 8 VGPRs).
// ---------------------------------------------------------------------------
typedef __bf16 bf16;
typedef __attribute__((ext_vector_type(16))) __bf16 v16bf;
typedef __attribute__((ext_vector_type(8)))  float  v8f;

union Frag16 {
    v16bf v;
    uint4 q[2];   // two 16-byte chunks
};

__device__ __forceinline__ bf16 f2bf(float f) {
    union { float f; uint32_t u; } v; v.f = f;
    uint32_t r = v.u + 0x7FFFu + ((v.u >> 16) & 1u);   // round-to-nearest-even
    uint16_t h = (uint16_t)(r >> 16);
    union { uint16_t u; bf16 b; } o; o.u = h;
    return o.b;
}

__device__ __forceinline__ float silu_f(float x) {
    return x / (1.0f + __expf(-x));
}

// ---------------------------------------------------------------------------
// fp32 -> bf16 pack
// ---------------------------------------------------------------------------
__global__ void pack_bf16_kernel(const float* __restrict__ src,
                                 bf16* __restrict__ dst, int n) {
    int i = blockIdx.x * blockDim.x + threadIdx.x;
    if (i < n) dst[i] = f2bf(src[i]);
}

// ---------------------------------------------------------------------------
// Generic bf16 WMMA GEMM:  C[M,N] (f32) = A[M,K] (bf16, row-major)
//                                       x W[N,K]^T (bf16, row-major rows = N)
// One wave computes a 32x32 tile (2x2 fragments of 16x16, K-step 32).
// M, N multiples of 32; K multiple of 32; rows 16B-aligned.
// ---------------------------------------------------------------------------
__global__ void gemm_bf16_wmma(const bf16* __restrict__ A,
                               const bf16* __restrict__ W,
                               float* __restrict__ C,
                               int M, int N, int K) {
    const int lane = threadIdx.x & 31;
    const int half = lane >> 4;        // 0: lanes 0-15, 1: lanes 16-31
    const int idx  = lane & 15;        // row (A) / col (B) within tile

    const int tilesN = N >> 5;
    const int tilesM = M >> 5;
    const int wave = blockIdx.x * (blockDim.x >> 5) + (threadIdx.x >> 5);
    if (wave >= tilesM * tilesN) return;   // wave-uniform: EXEC stays all-ones

    const int m0 = (wave / tilesN) << 5;
    const int n0 = (wave % tilesN) << 5;

    // Per-lane row bases; ISA 16-bit fragment layout:
    //   VGPR v<4 : K = half*8 + 2v .. +1     (16B chunk at k + half*8)
    //   VGPR v>=4: K = 16 + half*8 + 2(v-4) (16B chunk at k + 16 + half*8)
    const bf16* a0p = A + (size_t)(m0 + idx) * K + half * 8;
    const bf16* a1p = a0p + (size_t)16 * K;
    const bf16* b0p = W + (size_t)(n0 + idx) * K + half * 8;
    const bf16* b1p = b0p + (size_t)16 * K;

    v8f c00 = {}, c01 = {}, c10 = {}, c11 = {};

    for (int kk = 0; kk < K; kk += 32) {
        Frag16 a0, a1, b0, b1;
        a0.q[0] = *(const uint4*)(a0p + kk);
        a0.q[1] = *(const uint4*)(a0p + kk + 16);
        a1.q[0] = *(const uint4*)(a1p + kk);
        a1.q[1] = *(const uint4*)(a1p + kk + 16);
        b0.q[0] = *(const uint4*)(b0p + kk);
        b0.q[1] = *(const uint4*)(b0p + kk + 16);
        b1.q[0] = *(const uint4*)(b1p + kk);
        b1.q[1] = *(const uint4*)(b1p + kk + 16);

        __builtin_prefetch(a0p + kk + 128, 0, 0);
        __builtin_prefetch(b0p + kk + 128, 0, 0);

        c00 = __builtin_amdgcn_wmma_f32_16x16x32_bf16(false, a0.v, false, b0.v,
                                                      (short)0, c00, false, false);
        c01 = __builtin_amdgcn_wmma_f32_16x16x32_bf16(false, a0.v, false, b1.v,
                                                      (short)0, c01, false, false);
        c10 = __builtin_amdgcn_wmma_f32_16x16x32_bf16(false, a1.v, false, b0.v,
                                                      (short)0, c10, false, false);
        c11 = __builtin_amdgcn_wmma_f32_16x16x32_bf16(false, a1.v, false, b1.v,
                                                      (short)0, c11, false, false);
    }

    // C/D layout: VGPR r -> row half*8 + r, lane idx -> column
#pragma unroll
    for (int r = 0; r < 8; ++r) {
        const int row0 = m0 + half * 8 + r;
        const int row1 = row0 + 16;
        const int col0 = n0 + idx;
        const int col1 = col0 + 16;
        C[(size_t)row0 * N + col0] = c00[r];
        C[(size_t)row0 * N + col1] = c01[r];
        C[(size_t)row1 * N + col0] = c10[r];
        C[(size_t)row1 * N + col1] = c11[r];
    }
}

// ---------------------------------------------------------------------------
// Depthwise causal conv (K=4) + bias + SiLU over x = xz[:, :d_inner].
// Writes fp32 (for scan) and bf16 (for x_proj GEMM).
// ---------------------------------------------------------------------------
__global__ void conv_silu_kernel(const float* __restrict__ xz,
                                 const float* __restrict__ cw,   // [d_inner,4]
                                 const float* __restrict__ cb,   // [d_inner]
                                 float* __restrict__ xf,
                                 bf16*  __restrict__ xb,
                                 int L, int d_inner, int ld_xz, int total) {
    int i = blockIdx.x * blockDim.x + threadIdx.x;
    if (i >= total) return;
    const int d  = i % d_inner;
    const int ml = i / d_inner;          // b*L + l
    const int l  = ml % L;
    const int b  = ml / L;

    float acc = cb[d];
#pragma unroll
    for (int k = 0; k < 4; ++k) {
        const int ll = l + k - 3;        // left-pad of 3
        if (ll >= 0)
            acc += xz[(size_t)(b * L + ll) * ld_xz + d] * cw[d * 4 + k];
    }
    const float s = silu_f(acc);
    const size_t o = (size_t)ml * d_inner + d;
    xf[o] = s;
    xb[o] = f2bf(s);
}

// ---------------------------------------------------------------------------
// dt = softplus(x_dbl[:, :dt_rank] @ dt_proj_w^T + dt_proj_b)
// dt_proj_w is 512KB -> fully L2-resident across the 4M dot products.
// ---------------------------------------------------------------------------
__global__ void dt_softplus_kernel(const float* __restrict__ xdbl,  // [M, ld]
                                   const float* __restrict__ dtw,   // [d_inner, dt_rank]
                                   const float* __restrict__ dtb,   // [d_inner]
                                   float* __restrict__ dt,          // [M, d_inner]
                                   int Mrows, int d_inner, int dt_rank,
                                   int ld_xdbl) {
    int i = blockIdx.x * blockDim.x + threadIdx.x;
    if (i >= Mrows * d_inner) return;
    const int d = i % d_inner;
    const int m = i / d_inner;

    const float* __restrict__ xr = xdbl + (size_t)m * ld_xdbl;
    const float* __restrict__ wr = dtw + (size_t)d * dt_rank;
    float acc = dtb[d];
#pragma unroll 8
    for (int r = 0; r < 64; ++r) acc += xr[r] * wr[r];

    const float sp = (acc > 20.0f) ? acc : log1pf(__expf(acc));
    dt[(size_t)m * d_inner + d] = sp;
}

// ---------------------------------------------------------------------------
// Selective scan. One thread per (b, d): 16-state recurrence over L, state +
// A-row in registers. B/C rows for a 64-step chunk staged in LDS (shared by
// all 256 channels in the block). Fuses +x*D, *silu(z), and bf16 repack of y.
// ---------------------------------------------------------------------------
__global__ void scan_kernel(const float* __restrict__ dt,    // [M, d_inner]
                            const float* __restrict__ xf,    // [M, d_inner]
                            const float* __restrict__ xdbl,  // [M, 96] (B at 64, C at 80)
                            const float* __restrict__ xz,    // [M, 4096] (z at +d_inner)
                            const float* __restrict__ A_log, // [d_inner, 16]
                            const float* __restrict__ Dp,    // [d_inner]
                            bf16* __restrict__ yb,           // [M, d_inner]
                            int L, int d_inner, int ld_xdbl, int ld_xz) {
    const int b = blockIdx.x;
    const int d = blockIdx.y * blockDim.x + threadIdx.x;

    __shared__ float sBC[64][33];        // 64 steps x (16 B | 16 C), padded

    float Aa[16], h[16];
#pragma unroll
    for (int n = 0; n < 16; ++n) {
        Aa[n] = -__expf(A_log[(size_t)d * 16 + n]);
        h[n] = 0.0f;
    }
    const float Dv = Dp[d];

    for (int l0 = 0; l0 < L; l0 += 64) {
        __syncthreads();
        for (int e = threadIdx.x; e < 64 * 32; e += blockDim.x) {
            const int row = e >> 5, col = e & 31;
            sBC[row][col] = xdbl[(size_t)(b * L + l0 + row) * ld_xdbl + 64 + col];
        }
        __syncthreads();

        for (int t = 0; t < 64; ++t) {
            const size_t m = (size_t)(b * L + l0 + t);
            const float dtv = dt[m * d_inner + d];
            const float xv  = xf[m * d_inner + d];
            const float zv  = xz[m * ld_xz + d_inner + d];

            float y = 0.0f;
#pragma unroll
            for (int n = 0; n < 16; ++n) {
                const float dA = __expf(dtv * Aa[n]);
                h[n] = dA * h[n] + dtv * sBC[t][n] * xv;
                y += h[n] * sBC[t][16 + n];
            }
            y = (y + xv * Dv) * silu_f(zv);
            yb[m * d_inner + d] = f2bf(y);
        }
    }
}

// ---------------------------------------------------------------------------
// Host launcher
// ---------------------------------------------------------------------------
extern "C" void kernel_launch(void* const* d_in, const int* in_sizes, int n_in,
                              void* d_out, int out_size, void* d_ws, size_t ws_size,
                              hipStream_t stream) {
    const float* hidden     = (const float*)d_in[0];
    const float* in_proj_w  = (const float*)d_in[1];
    const float* conv_w     = (const float*)d_in[2];
    const float* conv_b     = (const float*)d_in[3];
    const float* x_proj_w   = (const float*)d_in[4];
    const float* dt_proj_w  = (const float*)d_in[5];
    const float* dt_proj_b  = (const float*)d_in[6];
    const float* A_log      = (const float*)d_in[7];
    const float* Dp         = (const float*)d_in[8];
    const float* out_proj_w = (const float*)d_in[9];

    const int Bsz = 2, L = 1024, d_model = 1024, d_inner = 2048, dt_rank = 64;
    const int Mrows   = Bsz * L;                 // 2048
    const int ld_xz   = 2 * d_inner;             // 4096
    const int ld_xdbl = dt_rank + 2 * 16;        // 96

    // workspace carve-out (~97 MB total), 256B aligned
    char* ws = (char*)d_ws;
    size_t cur = 0;
    auto alloc = [&](size_t bytes) -> void* {
        void* p = ws + cur;
        cur = (cur + bytes + 255) & ~(size_t)255;
        return p;
    };
    bf16*  hid_b = (bf16*) alloc((size_t)Mrows * d_model * sizeof(bf16));      // 4 MB
    bf16*  inw_b = (bf16*) alloc((size_t)2 * d_inner * d_model * sizeof(bf16));// 8 MB
    bf16*  xpw_b = (bf16*) alloc((size_t)ld_xdbl * d_inner * sizeof(bf16));    // 384 KB
    bf16*  opw_b = (bf16*) alloc((size_t)d_model * d_inner * sizeof(bf16));    // 4 MB
    float* xz    = (float*)alloc((size_t)Mrows * ld_xz * sizeof(float));       // 32 MB
    float* xf    = (float*)alloc((size_t)Mrows * d_inner * sizeof(float));     // 16 MB
    bf16*  xb    = (bf16*) alloc((size_t)Mrows * d_inner * sizeof(bf16));      // 8 MB
    float* xdbl  = (float*)alloc((size_t)Mrows * ld_xdbl * sizeof(float));     // 768 KB
    float* dtf   = (float*)alloc((size_t)Mrows * d_inner * sizeof(float));     // 16 MB
    bf16*  yb    = (bf16*) alloc((size_t)Mrows * d_inner * sizeof(bf16));      // 8 MB

    const int TPB = 256;
    auto nblk = [](size_t n, int t) { return (unsigned)((n + t - 1) / t); };

    // 1) pack weights / activations to bf16
    {
        size_t n;
        n = (size_t)Mrows * d_model;
        pack_bf16_kernel<<<nblk(n, TPB), TPB, 0, stream>>>(hidden, hid_b, (int)n);
        n = (size_t)2 * d_inner * d_model;
        pack_bf16_kernel<<<nblk(n, TPB), TPB, 0, stream>>>(in_proj_w, inw_b, (int)n);
        n = (size_t)ld_xdbl * d_inner;
        pack_bf16_kernel<<<nblk(n, TPB), TPB, 0, stream>>>(x_proj_w, xpw_b, (int)n);
        n = (size_t)d_model * d_inner;
        pack_bf16_kernel<<<nblk(n, TPB), TPB, 0, stream>>>(out_proj_w, opw_b, (int)n);
    }

    // 2) in_proj GEMM: xz[2048,4096] = hidden[2048,1024] @ in_proj_w[4096,1024]^T
    {
        const int M = Mrows, N = 2 * d_inner, K = d_model;
        const int waves = (M >> 5) * (N >> 5);
        gemm_bf16_wmma<<<nblk((size_t)waves * 32, TPB), TPB, 0, stream>>>(
            hid_b, inw_b, xz, M, N, K);
    }

    // 3) depthwise conv + SiLU
    {
        const size_t total = (size_t)Mrows * d_inner;
        conv_silu_kernel<<<nblk(total, TPB), TPB, 0, stream>>>(
            xz, conv_w, conv_b, xf, xb, L, d_inner, ld_xz, (int)total);
    }

    // 4) x_proj GEMM: xdbl[2048,96] = x[2048,2048] @ x_proj_w[96,2048]^T
    {
        const int M = Mrows, N = ld_xdbl, K = d_inner;
        const int waves = (M >> 5) * (N >> 5);
        gemm_bf16_wmma<<<nblk((size_t)waves * 32, TPB), TPB, 0, stream>>>(
            xb, xpw_b, xdbl, M, N, K);
    }

    // 5) dt projection + softplus
    {
        const size_t total = (size_t)Mrows * d_inner;
        dt_softplus_kernel<<<nblk(total, TPB), TPB, 0, stream>>>(
            xdbl, dt_proj_w, dt_proj_b, dtf, Mrows, d_inner, dt_rank, ld_xdbl);
    }

    // 6) selective scan (fused +x*D, *silu(z), bf16 pack)
    {
        dim3 grid(Bsz, d_inner / TPB);   // (2, 8)
        scan_kernel<<<grid, TPB, 0, stream>>>(
            dtf, xf, xdbl, xz, A_log, Dp, yb, L, d_inner, ld_xdbl, ld_xz);
    }

    // 7) out_proj GEMM: out[2048,1024] = y[2048,2048] @ out_proj_w[1024,2048]^T
    {
        const int M = Mrows, N = d_model, K = d_inner;
        const int waves = (M >> 5) * (N >> 5);
        gemm_bf16_wmma<<<nblk((size_t)waves * 32, TPB), TPB, 0, stream>>>(
            yb, opw_b, (float*)d_out, M, N, K);
    }
}
